// LSTM_23484881174800
// MI455X (gfx1250) — matrix operand order repaired and checked by reading, not compiled, not run on hardware
//
#include <hip/hip_runtime.h>
#include <hip/hip_bf16.h>

// ---------------------------------------------------------------------------
// Persistent 2-layer LSTM (HID=64) on CDNA5 WMMA (bf16 in, f32 accumulate).
// Block = 128 threads = 4 wave32; each block owns 16 batch rows for the whole
// sequence. Waves split the 256 gate columns 4-ways; h1/h2 are exchanged
// through LDS in WMMA A-operand layout each step. Cell state c stays in f32
// registers for all 1040 steps.
// ---------------------------------------------------------------------------

typedef __attribute__((ext_vector_type(16))) __bf16        v16bf;
typedef __attribute__((ext_vector_type(8)))  float         v8f;
typedef __attribute__((ext_vector_type(4)))  unsigned int  v4u;

#define HIDN 64
#define NFEAT 3
#define TSEQ 1024
#define FPRED 16
#define BATCH 8192
#define OUTW ((TSEQ + FPRED) * NFEAT)   // 3120

__device__ __forceinline__ unsigned short f2bf(float x) {
  unsigned int u = __float_as_uint(x);
  u += 0x7fffu + ((u >> 16) & 1u);      // round-to-nearest-even
  return (unsigned short)(u >> 16);
}
__device__ __forceinline__ float fsig(float x) {
  return __builtin_amdgcn_rcpf(1.0f + __expf(-x));
}
__device__ __forceinline__ float ftanh(float x) {
  float e = __expf(2.0f * x);
  return 1.0f - 2.0f * __builtin_amdgcn_rcpf(e + 1.0f);
}

union APack { v4u u[2]; v16bf v; };

// A operand (16x32 bf16): lane holds row M=lane&15; lanes<16 take K {0..15},
// lanes>=16 take K {16..31}  -> two adjacent 16B LDS chunks.
__device__ __forceinline__ v16bf loadA(const unsigned short* s, int row, int kb, int half) {
  const char* b = (const char*)s;
  int off = row * 128 + kb * 64 + half * 32;   // bytes; row stride = 64 halves
  APack p;
  p.u[0] = *(const v4u*)(b + off);
  p.u[1] = *(const v4u*)(b + off + 16);
  return p.v;
}
// B operand (32x16 bf16): lane holds col N=lane&15 (= weight row g), 16
// consecutive K values -> one contiguous 32B LDS chunk from row-major W[g][k].
__device__ __forceinline__ v16bf loadB(const unsigned short* s, int g, int kb, int half) {
  return *(const v16bf*)((const char*)s + g * 128 + kb * 64 + half * 32);
}

__global__ __launch_bounds__(128) void lstm2_persistent(
    const float* __restrict__ y,
    const float* __restrict__ Wih1, const float* __restrict__ bih1,
    const float* __restrict__ Whh1, const float* __restrict__ bhh1,
    const float* __restrict__ Wih2, const float* __restrict__ bih2,
    const float* __restrict__ Whh2, const float* __restrict__ bhh2,
    const float* __restrict__ Wlin, const float* __restrict__ blin,
    float* __restrict__ out)
{
  extern __shared__ unsigned char smem[];
  unsigned short* sWhh1 = (unsigned short*)smem;       // [256][64] bf16  32KB
  unsigned short* sWih2 = sWhh1 + 16384;               // [256][64] bf16  32KB
  unsigned short* sWhh2 = sWih2 + 16384;               // [256][64] bf16  32KB
  unsigned short* sWlin = sWhh2 + 16384;               // [16][64]  bf16   2KB (rows>=3 zero)
  unsigned short* sHA1  = sWlin + 1024;                // h1 in A layout   2KB
  unsigned short* sHA2  = sHA1 + 1024;                 // h2 in A layout   2KB
  float4*         sWih1b = (float4*)(sHA2 + 1024);     // [256] {w0,w1,w2,b1} 4KB
  float*          sBias2 = (float*)(sWih1b + 256);     // [256] b_ih2+b_hh2   1KB
  float*          sBlin  = sBias2 + 256;               // [16]
  float*          sX     = sBlin + 16;                 // [16][4] current x

  const int tid      = threadIdx.x;
  const int lane     = tid & 31;
  const int wave     = tid >> 5;
  const int laneN    = lane & 15;
  const int laneHalf = lane >> 4;
  const int hw       = wave * 16;          // this wave's 16 hidden columns
  const int m0       = blockIdx.x * 16;    // batch-row base of this block

  // ---- one-time weight staging (f32 -> bf16, B-operand friendly) ----------
  for (int i = tid; i < 16384; i += 128) {
    sWhh1[i] = f2bf(Whh1[i]);
    sWih2[i] = f2bf(Wih2[i]);
    sWhh2[i] = f2bf(Whh2[i]);
  }
  for (int i = tid; i < 256; i += 128) {
    sWih1b[i] = make_float4(Wih1[i * 3 + 0], Wih1[i * 3 + 1], Wih1[i * 3 + 2],
                            bih1[i] + bhh1[i]);
    sBias2[i] = bih2[i] + bhh2[i];
  }
  for (int i = tid; i < 1024; i += 128) {
    int n = i >> 6, k = i & 63;
    sWlin[i] = (n < NFEAT) ? f2bf(Wlin[n * HIDN + k]) : (unsigned short)0;
    sHA1[i] = 0; sHA2[i] = 0;
  }
  if (tid < 16) sBlin[tid] = (tid < NFEAT) ? blin[tid] : 0.0f;
  for (int i = tid; i < 64; i += 128) sX[i] = 0.0f;

  float c1r[8], c2r[8];
#pragma unroll
  for (int r = 0; r < 8; r++) { c1r[r] = 0.0f; c2r[r] = 0.0f; }

  __syncthreads();

  for (int t = 0; t < TSEQ + FPRED; t++) {
    // ---- stage x[t] (global) or keep autoregressive feedback in sX --------
    if (t < TSEQ && wave == 0 && lane < 16) {
      size_t base = (size_t)(m0 + lane) * (TSEQ * NFEAT) + (size_t)t * NFEAT;
      sX[lane * 4 + 0] = y[base + 0];
      sX[lane * 4 + 1] = y[base + 1];
      sX[lane * 4 + 2] = y[base + 2];
    }
    __syncthreads();                                      // sX + h tiles ready

    v16bf a1[2];
    a1[0] = loadA(sHA1, laneN, 0, laneHalf);
    a1[1] = loadA(sHA1, laneN, 1, laneHalf);
    float4 xr[8];
#pragma unroll
    for (int r = 0; r < 8; r++)
      xr[r] = *(const float4*)&sX[(r + 8 * laneHalf) * 4];

    // ---- layer 1: gates[16, hw..hw+15] for q = i,f,g,o --------------------
    v8f gacc[4];
#pragma unroll
    for (int q = 0; q < 4; q++) {
      const int g0 = q * 64 + hw;
      float4 w4 = sWih1b[g0 + laneN];
      v8f acc;
#pragma unroll
      for (int r = 0; r < 8; r++)
        acc[r] = w4.w + xr[r].x * w4.x + xr[r].y * w4.y + xr[r].z * w4.z;
#pragma unroll
      for (int kb = 0; kb < 2; kb++) {
        v16bf b = loadB(sWhh1, g0 + laneN, kb, laneHalf);
        acc = __builtin_amdgcn_wmma_f32_16x16x32_bf16(
            false, a1[kb], false, b, (short)0, acc, false, false);
      }
      gacc[q] = acc;
    }
    float h1v[8];
#pragma unroll
    for (int r = 0; r < 8; r++) {
      float i_ = fsig(gacc[0][r]);
      float f_ = fsig(gacc[1][r]);
      float g_ = ftanh(gacc[2][r]);
      float o_ = fsig(gacc[3][r]);
      float c = f_ * c1r[r] + i_ * g_;
      c1r[r] = c;
      h1v[r] = o_ * ftanh(c);
    }
    __syncthreads();                                      // done reading old h1
#pragma unroll
    for (int r = 0; r < 8; r++)
      sHA1[(r + 8 * laneHalf) * 64 + hw + laneN] = f2bf(h1v[r]);
    __syncthreads();                                      // new h1 visible

    // ---- layer 2: h1@Wih2' + h2@Whh2' -------------------------------------
    v16bf a1n[2], a2[2];
    a1n[0] = loadA(sHA1, laneN, 0, laneHalf);
    a1n[1] = loadA(sHA1, laneN, 1, laneHalf);
    a2[0]  = loadA(sHA2, laneN, 0, laneHalf);
    a2[1]  = loadA(sHA2, laneN, 1, laneHalf);
#pragma unroll
    for (int q = 0; q < 4; q++) {
      const int g0 = q * 64 + hw;
      float bz = sBias2[g0 + laneN];
      v8f acc;
#pragma unroll
      for (int r = 0; r < 8; r++) acc[r] = bz;
#pragma unroll
      for (int kb = 0; kb < 2; kb++) {
        v16bf b1 = loadB(sWih2, g0 + laneN, kb, laneHalf);
        acc = __builtin_amdgcn_wmma_f32_16x16x32_bf16(
            false, a1n[kb], false, b1, (short)0, acc, false, false);
        v16bf b2 = loadB(sWhh2, g0 + laneN, kb, laneHalf);
        acc = __builtin_amdgcn_wmma_f32_16x16x32_bf16(
            false, a2[kb], false, b2, (short)0, acc, false, false);
      }
      gacc[q] = acc;
    }
    float h2v[8];
#pragma unroll
    for (int r = 0; r < 8; r++) {
      float i_ = fsig(gacc[0][r]);
      float f_ = fsig(gacc[1][r]);
      float g_ = ftanh(gacc[2][r]);
      float o_ = fsig(gacc[3][r]);
      float c = f_ * c2r[r] + i_ * g_;
      c2r[r] = c;
      h2v[r] = o_ * ftanh(c);
    }
    __syncthreads();                                      // done reading old h2
#pragma unroll
    for (int r = 0; r < 8; r++)
      sHA2[(r + 8 * laneHalf) * 64 + hw + laneN] = f2bf(h2v[r]);
    __syncthreads();                                      // new h2 visible

    // ---- head: out = h2 @ Wlin' + blin (wave 0 only, uniform per wave) ----
    if (wave == 0) {
      v16bf aa0 = loadA(sHA2, laneN, 0, laneHalf);
      v16bf aa1 = loadA(sHA2, laneN, 1, laneHalf);
      float bl = sBlin[laneN];
      v8f acc;
#pragma unroll
      for (int r = 0; r < 8; r++) acc[r] = bl;
      acc = __builtin_amdgcn_wmma_f32_16x16x32_bf16(
          false, aa0, false, loadB(sWlin, laneN, 0, laneHalf), (short)0, acc, false, false);
      acc = __builtin_amdgcn_wmma_f32_16x16x32_bf16(
          false, aa1, false, loadB(sWlin, laneN, 1, laneHalf), (short)0, acc, false, false);
      if (laneN < NFEAT) {
#pragma unroll
        for (int r = 0; r < 8; r++) {
          int m = r + 8 * laneHalf;
          float v = acc[r];
          out[(size_t)(m0 + m) * OUTW + (size_t)t * NFEAT + laneN] = v;
          if (t >= TSEQ - 1) sX[m * 4 + laneN] = v;       // autoregressive feed
        }
      }
    }
    // next-iteration top barrier orders sX feedback vs. readers
  }
}

extern "C" void kernel_launch(void* const* d_in, const int* in_sizes, int n_in,
                              void* d_out, int out_size, void* d_ws, size_t ws_size,
                              hipStream_t stream) {
  (void)in_sizes; (void)n_in; (void)out_size; (void)d_ws; (void)ws_size;
  const float* y    = (const float*)d_in[0];
  const float* Wih1 = (const float*)d_in[1];
  const float* bih1 = (const float*)d_in[2];
  const float* Whh1 = (const float*)d_in[3];
  const float* bhh1 = (const float*)d_in[4];
  const float* Wih2 = (const float*)d_in[5];
  const float* bih2 = (const float*)d_in[6];
  const float* Whh2 = (const float*)d_in[7];
  const float* bhh2 = (const float*)d_in[8];
  const float* Wlin = (const float*)d_in[9];
  const float* blin = (const float*)d_in[10];
  float* out = (float*)d_out;

  const size_t shmem = 109888;   // 3x32KB bf16 weights + tiles/buffers
  hipFuncSetAttribute((const void*)lstm2_persistent,
                      hipFuncAttributeMaxDynamicSharedMemorySize, (int)shmem);
  lstm2_persistent<<<BATCH / 16, 128, shmem, stream>>>(
      y, Wih1, bih1, Whh1, bhh1, Wih2, bih2, Whh2, bhh2, Wlin, blin, out);
}
